// QuantizeEMAReset_11098195493289
// MI455X (gfx1250) — matrix operand order repaired and checked by reading, not compiled, not run on hardware
//
#include <hip/hip_runtime.h>

// ---------------- problem constants ----------------
#define DDIM    512
#define NBCODE  1024
#define TDIM    2048
#define NBATCH  32
#define MTOT    (NBATCH * TDIM)            // 65536 rows
#define XTOTAL  ((long)MTOT * DDIM)        // 33,554,432 elements
#define EPSF    1e-7f

// ---------------- tiling ----------------
#define ROWS_WG   128                       // rows (t-positions) per workgroup
#define ASTRIDE   520                       // LDS row stride in bf16 (16B aligned, de-conflicted)
#define ARGMAX_THREADS 128                  // 4 waves x 32 rows each
#define GATHER_BLOCKS 8192

// ---------------- workspace layout (bytes) ----------------
#define WS_CBH    0                         // 1024*512 bf16      = 1,048,576
#define WS_CNORM  1048576                   // 1024 f32           = 4,096
#define WS_IDX    1052672                   // 65536 u32          = 262,144
#define WS_HIST   1314816                   // 1024 u32           = 4,096
#define WS_PART   1318912                   // 8192 f32           = 32,768  (total 1,351,680)

// ---------------- output layout (floats) ----------------
#define OUT_X       0L
#define OUT_COMMIT  33554432L
#define OUT_PERP    33554433L
#define OUT_IDX     33554434L

typedef __attribute__((ext_vector_type(16))) __bf16 v16bf;
typedef __attribute__((ext_vector_type(8)))  float  v8f;

// ============ kernel 0: zero histogram ============
__global__ void vq_zero_kernel(unsigned* __restrict__ hist) {
    int i = blockIdx.x * blockDim.x + threadIdx.x;
    if (i < NBCODE) hist[i] = 0u;
}

// ============ kernel 1: codebook -> bf16, 0.5*||c||^2 ============
__global__ __launch_bounds__(256)
void vq_prep_kernel(const float* __restrict__ cb,
                    __bf16* __restrict__ cbh,
                    float* __restrict__ cnormh) {
    __shared__ float red[256];
    const int k = blockIdx.x;
    const float* row = cb + (long)k * DDIM;
    float s = 0.f;
    for (int d = threadIdx.x; d < DDIM; d += 256) {
        float v = row[d];
        cbh[(long)k * DDIM + d] = (__bf16)v;
        s += v * v;
    }
    red[threadIdx.x] = s;
    __syncthreads();
    for (int off = 128; off > 0; off >>= 1) {
        if (threadIdx.x < off) red[threadIdx.x] += red[threadIdx.x + off];
        __syncthreads();
    }
    if (threadIdx.x == 0) cnormh[k] = 0.5f * red[0];
}

// ============ kernel 2: WMMA argmax over codes ============
// score[m,k] = x[m].c[k] - 0.5*||c[k]||^2 ; argmax_k score == argmin_k dist
// 4 waves/WG, each wave owns 32 rows (two A fragments) so every B (codebook)
// fragment feeds two WMMAs: steady state = 8 wmma : 12 (8 vmem + 4 lds) loads.
__global__ __launch_bounds__(ARGMAX_THREADS)
void vq_argmax_kernel(const float* __restrict__ x,
                      const __bf16* __restrict__ cbh,
                      const float* __restrict__ cnormh,
                      unsigned* __restrict__ idx_out,
                      unsigned* __restrict__ hist) {
    __shared__ __bf16 As[ROWS_WG * ASTRIDE];   // 128 rows x 512 d (stride 520) bf16 = 130 KB

    const int tid   = threadIdx.x;
    const int wave  = tid >> 5;      // 0..3
    const int lane  = tid & 31;
    const int lhalf = lane >> 4;     // 0: lanes 0-15, 1: lanes 16-31
    const int l16   = lane & 15;

    const long mbase = (long)blockIdx.x * ROWS_WG;   // 128 | 2048 -> single n per WG
    const int  n     = (int)(mbase / TDIM);
    const int  tbase = (int)(mbase % TDIM);

    // ---- stage A tile: x[n][d][tbase+r] -> As[r][d] (bf16), coalesced along t ----
    for (int e = tid; e < ROWS_WG * DDIM; e += ARGMAX_THREADS) {
        int d = e >> 7;              // e / 128
        int r = e & 127;
        float v = x[((long)n * DDIM + d) * TDIM + tbase + r];
        As[r * ASTRIDE + d] = (__bf16)v;
    }
    __syncthreads();

    const int row0 = wave * 32;      // this wave's 32 rows within the tile

    // running best per lane: group g (g=0: rows row0+0..15, g=1: rows row0+16..31),
    // VGPR i -> row row0 + g*16 + i + lhalf*8, codes restricted to (code%16 == l16).
    float bestv[2][8];
    int   besti[2][8];
#pragma unroll
    for (int g = 0; g < 2; ++g)
#pragma unroll
        for (int i = 0; i < 8; ++i) { bestv[g][i] = -3.4e38f; besti[g][i] = 0; }

    // A-frag source rows for this lane (frag row M = l16), K-half offset per lane group
    const int kofs = (lhalf ? 16 : 0);
    const __bf16* arow0 = &As[(row0 + l16) * ASTRIDE + kofs];
    const __bf16* arow1 = &As[(row0 + 16 + l16) * ASTRIDE + kofs];

    for (int kb = 0; kb < NBCODE; kb += 64) {
        v8f acc0[4] = {};            // rows row0+0..15  x 64 codes
        v8f acc1[4] = {};            // rows row0+16..31 x 64 codes
#pragma unroll 1
        for (int db = 0; db < DDIM; db += 32) {
            // A fragments (16x32 bf16): lanes 0-15 K=0..15, lanes 16-31 K=16..31
            v16bf a0, a1;
            {
                const __bf16* p0 = arow0 + db;
                const __bf16* p1 = arow1 + db;
#pragma unroll
                for (int j = 0; j < 8; ++j) {
                    a0[j] = p0[j]; a0[8 + j] = p0[8 + j];
                    a1[j] = p1[j]; a1[8 + j] = p1[8 + j];
                }
            }
#pragma unroll
            for (int f = 0; f < 4; ++f) {
                // B fragment (32x16 bf16): N = l16, lanes 0-15 K=0..15, lanes 16-31 K=16..31
                const __bf16* bp = cbh + ((long)(kb + f * 16 + l16) << 9) + db + (lhalf << 4);
                v16bf b;
#pragma unroll
                for (int j = 0; j < 16; ++j) b[j] = bp[j];
                acc0[f] = __builtin_amdgcn_wmma_f32_16x16x32_bf16(
                    false, a0, false, b, (short)0, acc0[f], false, false);
                acc1[f] = __builtin_amdgcn_wmma_f32_16x16x32_bf16(
                    false, a1, false, b, (short)0, acc1[f], false, false);
            }
        }
        // fold bias, update running argmax (strict '>' keeps earliest code on ties)
#pragma unroll
        for (int f = 0; f < 4; ++f) {
            int   code = kb + f * 16 + l16;
            float bias = cnormh[code];
#pragma unroll
            for (int i = 0; i < 8; ++i) {
                float s0 = acc0[f][i] - bias;
                if (s0 > bestv[0][i]) { bestv[0][i] = s0; besti[0][i] = code; }
                float s1 = acc1[f][i] - bias;
                if (s1 > bestv[1][i]) { bestv[1][i] = s1; besti[1][i] = code; }
            }
        }
    }

    // cross-lane max-reduce over the 16 lanes sharing each row (halves are disjoint rows)
#pragma unroll
    for (int off = 1; off < 16; off <<= 1) {
#pragma unroll
        for (int g = 0; g < 2; ++g)
#pragma unroll
            for (int i = 0; i < 8; ++i) {
                float ov = __shfl_xor(bestv[g][i], off, 32);
                int   oi = __shfl_xor(besti[g][i], off, 32);
                if (ov > bestv[g][i] || (ov == bestv[g][i] && oi < besti[g][i])) {
                    bestv[g][i] = ov; besti[g][i] = oi;
                }
            }
    }
    if (l16 == 0) {                   // lanes 0 and 16 own rows +0..7 / +8..15 of each group
#pragma unroll
        for (int g = 0; g < 2; ++g)
#pragma unroll
            for (int i = 0; i < 8; ++i) {
                long m = mbase + row0 + g * 16 + lhalf * 8 + i;
                idx_out[m] = (unsigned)besti[g][i];
                atomicAdd(&hist[besti[g][i]], 1u);
            }
    }
}

// ============ kernel 3: gather x_d, write x_out, commitment partials ============
__global__ __launch_bounds__(256)
void vq_gather_kernel(const float* __restrict__ x,
                      const float* __restrict__ cb,
                      const unsigned* __restrict__ idx,
                      float* __restrict__ xout,
                      float* __restrict__ partials) {
    __shared__ float red[256];
    float s = 0.f;
    const long stride = (long)GATHER_BLOCKS * 256;
    for (long e = (long)blockIdx.x * 256 + threadIdx.x; e < XTOTAL; e += stride) {
        long nt = e / TDIM;          // n*512 + d
        int  t  = (int)(e % TDIM);
        int  d  = (int)(nt % DDIM);
        int  n  = (int)(nt / DDIM);
        unsigned code = idx[(long)n * TDIM + t];
        float xd = cb[((long)code << 9) + d];   // f32 codebook, L2-resident
        float xv = x[e];
        xout[e] = xd;                           // straight-through value == x_d
        float diff = xv - xd;
        s += diff * diff;
    }
    red[threadIdx.x] = s;
    __syncthreads();
    for (int off = 128; off > 0; off >>= 1) {
        if (threadIdx.x < off) red[threadIdx.x] += red[threadIdx.x + off];
        __syncthreads();
    }
    if (threadIdx.x == 0) partials[blockIdx.x] = red[0];
}

// ============ kernel 4: indices -> float output ============
__global__ void vq_idxf_kernel(const unsigned* __restrict__ idx, float* __restrict__ outf) {
    int m = blockIdx.x * 256 + threadIdx.x;
    if (m < MTOT) outf[m] = (float)idx[m];
}

// ============ kernel 5: scalars (commitment, perplexity) ============
__global__ __launch_bounds__(256)
void vq_scalar_kernel(const unsigned* __restrict__ hist,
                      const float* __restrict__ partials,
                      float* __restrict__ out_commit,
                      float* __restrict__ out_perp) {
    __shared__ float red[256];
    // commitment: fixed-order sum of per-block partials (deterministic)
    float cs = 0.f;
    for (int i = threadIdx.x; i < GATHER_BLOCKS; i += 256) cs += partials[i];
    red[threadIdx.x] = cs;
    __syncthreads();
    for (int off = 128; off > 0; off >>= 1) {
        if (threadIdx.x < off) red[threadIdx.x] += red[threadIdx.x + off];
        __syncthreads();
    }
    float commit = red[0] / (float)XTOTAL;
    __syncthreads();
    // perplexity: exp(-sum(p*log(p+eps)))
    float es = 0.f;
    for (int k = threadIdx.x; k < NBCODE; k += 256) {
        float p = (float)hist[k] / (float)MTOT;
        es += p * logf(p + EPSF);
    }
    red[threadIdx.x] = es;
    __syncthreads();
    for (int off = 128; off > 0; off >>= 1) {
        if (threadIdx.x < off) red[threadIdx.x] += red[threadIdx.x + off];
        __syncthreads();
    }
    if (threadIdx.x == 0) {
        *out_commit = commit;
        *out_perp   = expf(-red[0]);
    }
}

// ============ host entry ============
extern "C" void kernel_launch(void* const* d_in, const int* in_sizes, int n_in,
                              void* d_out, int out_size, void* d_ws, size_t ws_size,
                              hipStream_t stream) {
    (void)in_sizes; (void)n_in; (void)out_size; (void)ws_size;
    const float* x  = (const float*)d_in[0];   // (32, 512, 2048) f32
    const float* cb = (const float*)d_in[1];   // (1024, 512) f32
    float* out = (float*)d_out;

    char* ws = (char*)d_ws;
    __bf16*   cbh      = (__bf16*)(ws + WS_CBH);
    float*    cnormh   = (float*)(ws + WS_CNORM);
    unsigned* idx      = (unsigned*)(ws + WS_IDX);
    unsigned* hist     = (unsigned*)(ws + WS_HIST);
    float*    partials = (float*)(ws + WS_PART);

    float* xout       = out + OUT_X;
    float* out_commit = out + OUT_COMMIT;
    float* out_perp   = out + OUT_PERP;
    float* out_idx    = out + OUT_IDX;

    vq_zero_kernel  <<<4, 256, 0, stream>>>(hist);
    vq_prep_kernel  <<<NBCODE, 256, 0, stream>>>(cb, cbh, cnormh);
    vq_argmax_kernel<<<MTOT / ROWS_WG, ARGMAX_THREADS, 0, stream>>>(x, cbh, cnormh, idx, hist);
    vq_gather_kernel<<<GATHER_BLOCKS, 256, 0, stream>>>(x, cb, idx, xout, partials);
    vq_idxf_kernel  <<<MTOT / 256, 256, 0, stream>>>(idx, out_idx);
    vq_scalar_kernel<<<1, 256, 0, stream>>>(hist, partials, out_commit, out_perp);
}
